// RingAttentionLlama_58128087384950
// MI455X (gfx1250) — compile-verified, matrix operand order checked by reading, not run on hardware
//
#include <hip/hip_runtime.h>
#include <hip/hip_bf16.h>

#define B_ 2
#define S_ 2048
#define D_ 2048
#define H_ 16
#define KVH_ 4
#define HD_ 128

typedef __attribute__((ext_vector_type(16))) __bf16 v16bf;
typedef __attribute__((ext_vector_type(8)))  float  v8f;

// ---------------------------------------------------------------------------
// WMMA helpers.  16-bit A-matrix 16x32 layout (ISA 7.12.2):
//   lanes 0-15 : row M = lane   ; elems 0..7 -> K=0..7,  elems 8..15 -> K=16..23
//   lanes16-31 : row M = lane-16; elems 0..7 -> K=8..15, elems 8..15 -> K=24..31
// With row-major (contiguous-K) storage this is two 16B loads per lane.
// B operand mirrors A with N in place of M, loaded from BT[n][k].
// ---------------------------------------------------------------------------
__device__ __forceinline__ v16bf load_frag16(const __bf16* rowptr, int k0, int half) {
  union { v16bf v; uint4 q[2]; } f;
  const char* p = (const char*)(rowptr + k0);
  f.q[0] = *(const uint4*)(p + half * 16);
  f.q[1] = *(const uint4*)(p + 32 + half * 16);
  return f.v;
}

__device__ __forceinline__ v8f wmma_bf16(v16bf a, v16bf b, v8f c) {
  return __builtin_amdgcn_wmma_f32_16x16x32_bf16(false, a, false, b, (short)0, c,
                                                 false, false);
}

// ---------------------------------------------------------------------------
// CDNA5 async copy: global -> LDS, 16 bytes per lane, tracked by ASYNCcnt.
// (ISA 15.18.3 GLOBAL_LOAD_ASYNC_TO_LDS_B128; VDST = LDS byte address VGPR.)
// Generic pointers to __shared__ carry the LDS byte offset in their low 32
// bits (ISA 10.2: LDS aperture truncates to addr[31:0]).
// ---------------------------------------------------------------------------
__device__ __forceinline__ void async_copy_b128(const __bf16* lds_dst,
                                                const __bf16* gsrc) {
  unsigned int loff = (unsigned int)(uintptr_t)lds_dst;
  unsigned long long gaddr = (unsigned long long)(uintptr_t)gsrc;
  asm volatile("global_load_async_to_lds_b128 %0, %1, off"
               :
               : "v"(loff), "v"(gaddr)
               : "memory");
}

__device__ __forceinline__ void wait_async0() {
  asm volatile("s_wait_asynccnt 0x0" ::: "memory");
}

// ---------------------------------------------------------------------------
// Elementwise f32 -> bf16 cast
// ---------------------------------------------------------------------------
__global__ void cast_bf16_kernel(const float* __restrict__ in,
                                 __bf16* __restrict__ out, int n) {
  int i = blockIdx.x * 256 + threadIdx.x;
  if (i < n) out[i] = (__bf16)in[i];
}

// ---------------------------------------------------------------------------
// Cast + transpose: B (K x N, f32, row-major) -> BT (N x K, bf16, row-major)
// grid = (N/32, K/32), block = (32, 8)
// ---------------------------------------------------------------------------
__global__ void transpose_cast_kernel(const float* __restrict__ Bm,
                                      __bf16* __restrict__ BT, int K, int N) {
  __shared__ float tile[32][33];
  int bx = blockIdx.x * 32;  // n
  int by = blockIdx.y * 32;  // k
  int tx = threadIdx.x, ty = threadIdx.y;
#pragma unroll
  for (int i = 0; i < 32; i += 8)
    tile[ty + i][tx] = Bm[(size_t)(by + ty + i) * N + bx + tx];
  __syncthreads();
#pragma unroll
  for (int i = 0; i < 32; i += 8)
    BT[(size_t)(bx + ty + i) * K + by + tx] = (__bf16)tile[tx][ty + i];
}

// ---------------------------------------------------------------------------
// Async-staged WMMA GEMM: C[M x N] = A[M x K] * B[K x N]
// A bf16 row-major, B pre-transposed as BT[N x K] bf16 row-major.
// Block = 256 threads (8 waves); tile BM=128 x BN=64 x BK=64,
// double-buffered LDS filled with global_load_async_to_lds_b128.
// Wave w computes rows [w*16, w*16+16) x the full 64-wide N strip.
// grid = (M/128, N/64).
// ---------------------------------------------------------------------------
#define LDSROW 72  // bf16 elements per LDS tile row (64 data + 8 pad = 144 B)

__device__ __forceinline__ void stage_tiles_async(
    const __bf16* __restrict__ A, const __bf16* __restrict__ BT, int K, int m0,
    int n0, int k0, __bf16* as, __bf16* bs, int tid) {
  // A tile: 128 rows x 64 cols -> 1024 x 16B chunks, 4 per thread
#pragma unroll
  for (int i = 0; i < 4; ++i) {
    int id = tid + i * 256;
    int seg = id & 7, row = id >> 3;
    async_copy_b128(as + row * LDSROW + seg * 8,
                    A + (size_t)(m0 + row) * K + k0 + seg * 8);
  }
  // B tile: 64 rows x 64 cols -> 512 x 16B chunks, 2 per thread
#pragma unroll
  for (int i = 0; i < 2; ++i) {
    int id = tid + i * 256;
    int seg = id & 7, row = id >> 3;
    async_copy_b128(bs + row * LDSROW + seg * 8,
                    BT + (size_t)(n0 + row) * K + k0 + seg * 8);
  }
}

template <bool F32OUT>
__global__ __launch_bounds__(256) void gemm_bt_kernel(
    const __bf16* __restrict__ A, const __bf16* __restrict__ BT,
    void* __restrict__ Cout, int M, int N, int K) {
  __shared__ __bf16 AsBuf[2][128 * LDSROW];
  __shared__ __bf16 BsBuf[2][64 * LDSROW];

  int tid = threadIdx.x;
  int w = tid >> 5, lane = tid & 31;
  int half = lane >> 4, ln = lane & 15;
  int m0 = blockIdx.x * 128;
  int n0 = blockIdx.y * 64;

  v8f acc0 = {0, 0, 0, 0, 0, 0, 0, 0};
  v8f acc1 = acc0, acc2 = acc0, acc3 = acc0;

  int nstages = K >> 6;  // BK = 64
  stage_tiles_async(A, BT, K, m0, n0, 0, AsBuf[0], BsBuf[0], tid);

  for (int s = 0; s < nstages; ++s) {
    wait_async0();       // this wave's async fills complete
    __syncthreads();     // all waves' fills complete
    int cur = s & 1;
    if (s + 1 < nstages)
      stage_tiles_async(A, BT, K, m0, n0, (s + 1) << 6, AsBuf[cur ^ 1],
                        BsBuf[cur ^ 1], tid);

    const __bf16* arow = AsBuf[cur] + (w * 16 + ln) * LDSROW;
    const __bf16* brow = BsBuf[cur] + ln * LDSROW;
#pragma unroll
    for (int kc = 0; kc < 2; ++kc) {
      v16bf a  = load_frag16(arow, kc * 32, half);
      v16bf b0 = load_frag16(brow, kc * 32, half);
      v16bf b1 = load_frag16(brow + 16 * LDSROW, kc * 32, half);
      v16bf b2 = load_frag16(brow + 32 * LDSROW, kc * 32, half);
      v16bf b3 = load_frag16(brow + 48 * LDSROW, kc * 32, half);
      acc0 = wmma_bf16(a, b0, acc0);
      acc1 = wmma_bf16(a, b1, acc1);
      acc2 = wmma_bf16(a, b2, acc2);
      acc3 = wmma_bf16(a, b3, acc3);
    }
    __syncthreads();     // reads done before buffer is overwritten
  }

  int rowbase = m0 + w * 16 + half * 8;  // C/D: elem r -> M = r + 8*half, N = ln
#pragma unroll
  for (int r = 0; r < 8; ++r) {
    size_t row = (size_t)(rowbase + r) * N;
    if (F32OUT) {
      float* C = (float*)Cout;
      C[row + n0 + ln]      = acc0[r];
      C[row + n0 + 16 + ln] = acc1[r];
      C[row + n0 + 32 + ln] = acc2[r];
      C[row + n0 + 48 + ln] = acc3[r];
    } else {
      __bf16* C = (__bf16*)Cout;
      C[row + n0 + ln]      = (__bf16)acc0[r];
      C[row + n0 + 16 + ln] = (__bf16)acc1[r];
      C[row + n0 + 32 + ln] = (__bf16)acc2[r];
      C[row + n0 + 48 + ln] = (__bf16)acc3[r];
    }
  }
}

// ---------------------------------------------------------------------------
// RoPE + reformat: in f32 [(b*S+s)][heads*128] -> out bf16 [b][h][s][128]
// ---------------------------------------------------------------------------
__global__ void rope_reformat_kernel(const float* __restrict__ inF,
                                     const float* __restrict__ cosT,
                                     const float* __restrict__ sinT,
                                     __bf16* __restrict__ outB, int heads,
                                     int total) {
  int idx = blockIdx.x * 256 + threadIdx.x;
  if (idx >= total) return;
  int i = idx & 63;
  int t = idx >> 6;
  int h = t % heads;
  int t2 = t / heads;
  int s = t2 % S_;
  int b = t2 / S_;
  size_t rowin = (size_t)b * S_ + s;
  const float* p = inF + rowin * ((size_t)heads * 128) + h * 128 + 2 * i;
  float x0 = p[0], x1 = p[1];
  float c = cosT[rowin * 64 + i];
  float sn = sinT[rowin * 64 + i];
  float o0 = x0 * c - x1 * sn;
  float o1 = x0 * sn + x1 * c;
  __bf16* q = outB + (((size_t)(b * heads + h) * S_ + s) * 128 + 2 * i);
  q[0] = (__bf16)o0;
  q[1] = (__bf16)o1;
}

// ---------------------------------------------------------------------------
// V reformat: f32 [(b*S+s)][g*128+hd] -> bf16 V^T layout [b][g][hd][s]
// ---------------------------------------------------------------------------
__global__ void vtrans_kernel(const float* __restrict__ vf,
                              __bf16* __restrict__ vb, int total) {
  int idx = blockIdx.x * 256 + threadIdx.x;
  if (idx >= total) return;
  int hd = idx & 127;
  int t = idx >> 7;
  int g = t % KVH_;
  int t2 = t / KVH_;
  int s = t2 % S_;
  int b = t2 / S_;
  vb[((size_t)(b * KVH_ + g) * 128 + hd) * S_ + s] =
      (__bf16)vf[((size_t)b * S_ + s) * (KVH_ * HD_) + g * 128 + hd];
}

// ---------------------------------------------------------------------------
// Causal GQA flash attention.
// qb: [b][h][s][128] bf16 (post-RoPE), kb: [b][g][s][128] bf16 (post-RoPE),
// vtb: [b][g][hd][s] bf16.  ao out: [(b*S+s)][h*128+hd] bf16.
// block = 128 threads (4 waves); each wave owns 16 query rows.
// grid = (S/64, B*H)
// ---------------------------------------------------------------------------
__global__ __launch_bounds__(128) void attn_kernel(
    const __bf16* __restrict__ qb, const __bf16* __restrict__ kb,
    const __bf16* __restrict__ vtb, __bf16* __restrict__ ao) {
  __shared__ __bf16 plds[4][16][32];  // per-wave P tile staging

  int w = threadIdx.x >> 5, lane = threadIdx.x & 31;
  int half = lane >> 4, ln = lane & 15;
  int qbase = blockIdx.x * 64 + w * 16;
  int bh = blockIdx.y;
  int b = bh >> 4, h = bh & 15;
  int g = h >> 2;  // GQA: rep = 4

  const __bf16* Qp = qb + (size_t)(b * H_ + h) * S_ * HD_;
  const __bf16* Kp = kb + (size_t)(b * KVH_ + g) * S_ * HD_;
  const __bf16* Vt = vtb + (size_t)(b * KVH_ + g) * HD_ * S_;

  v16bf aq[4];
  const __bf16* qrow = Qp + (size_t)(qbase + ln) * HD_;
#pragma unroll
  for (int c = 0; c < 4; ++c) aq[c] = load_frag16(qrow, c * 32, half);

  v8f zero = {0, 0, 0, 0, 0, 0, 0, 0};
  v8f o[8];
#pragma unroll
  for (int c = 0; c < 8; ++c) o[c] = zero;
  float mrow[8], lrow[8];
#pragma unroll
  for (int r = 0; r < 8; ++r) { mrow[r] = -1e30f; lrow[r] = 0.0f; }

  const float SCALE = 0.08838834764831845f;  // 1/sqrt(128)
  int myq = qbase + half * 8;                // elem r -> query row myq + r
  int nkb = (qbase + 47) >> 5;               // key blocks of 32 (causal bound)

  for (int kbi = 0; kbi < nkb; ++kbi) {
    int kb0 = kbi * 32;
    v8f s0 = zero, s1 = zero;
    const __bf16* krow0 = Kp + (size_t)(kb0 + ln) * HD_;
    const __bf16* krow1 = krow0 + (size_t)16 * HD_;
#pragma unroll
    for (int c = 0; c < 4; ++c) {
      v16bf bk0 = load_frag16(krow0, c * 32, half);
      v16bf bk1 = load_frag16(krow1, c * 32, half);
      s0 = wmma_bf16(aq[c], bk0, s0);
      s1 = wmma_bf16(aq[c], bk1, s1);
    }

    int key0 = kb0 + ln, key1 = key0 + 16;
#pragma unroll
    for (int r = 0; r < 8; ++r) {
      int qi = myq + r;
      float v0 = (key0 <= qi) ? s0[r] * SCALE : -1e30f;
      float v1 = (key1 <= qi) ? s1[r] * SCALE : -1e30f;
      float mx = fmaxf(v0, v1);
      mx = fmaxf(mx, __shfl_xor(mx, 1, 32));
      mx = fmaxf(mx, __shfl_xor(mx, 2, 32));
      mx = fmaxf(mx, __shfl_xor(mx, 4, 32));
      mx = fmaxf(mx, __shfl_xor(mx, 8, 32));
      float mnew = fmaxf(mrow[r], mx);
      float alpha = __expf(mrow[r] - mnew);
      float p0 = __expf(v0 - mnew);
      float p1 = __expf(v1 - mnew);
      float sum = p0 + p1;
      sum += __shfl_xor(sum, 1, 32);
      sum += __shfl_xor(sum, 2, 32);
      sum += __shfl_xor(sum, 4, 32);
      sum += __shfl_xor(sum, 8, 32);
      lrow[r] = lrow[r] * alpha + sum;
      mrow[r] = mnew;
#pragma unroll
      for (int c = 0; c < 8; ++c) o[c][r] *= alpha;
      int mq = r + half * 8;  // C-layout row -> P row
      plds[w][mq][ln] = (__bf16)p0;
      plds[w][mq][ln + 16] = (__bf16)p1;
    }

    v16bf ap = load_frag16(&plds[w][ln][0], 0, half);
#pragma unroll
    for (int c = 0; c < 8; ++c) {
      v16bf bv = load_frag16(Vt + (size_t)(c * 16 + ln) * S_, kb0, half);
      o[c] = wmma_bf16(ap, bv, o[c]);
    }
  }

#pragma unroll
  for (int c = 0; c < 8; ++c) {
#pragma unroll
    for (int r = 0; r < 8; ++r) {
      int row = myq + r;
      float val = o[c][r] / lrow[r];
      ao[((size_t)b * S_ + row) * D_ + h * 128 + c * 16 + ln] = (__bf16)val;
    }
  }
}

// ---------------------------------------------------------------------------
// launch
// ---------------------------------------------------------------------------
extern "C" void kernel_launch(void* const* d_in, const int* in_sizes, int n_in,
                              void* d_out, int out_size, void* d_ws,
                              size_t ws_size, hipStream_t stream) {
  const float* x    = (const float*)d_in[0];
  const float* cosT = (const float*)d_in[1];
  const float* sinT = (const float*)d_in[2];
  const float* wq   = (const float*)d_in[3];
  const float* wk   = (const float*)d_in[4];
  const float* wv   = (const float*)d_in[5];
  const float* wo   = (const float*)d_in[6];
  (void)in_sizes; (void)n_in; (void)out_size; (void)ws_size;

  const size_t MB = 1024 * 1024;
  char* ws = (char*)d_ws;
  __bf16* xb  = (__bf16*)(ws + 0);          // 16 MiB
  __bf16* wqT = (__bf16*)(ws + 16 * MB);    //  8 MiB
  __bf16* wkT = (__bf16*)(ws + 24 * MB);    //  2 MiB
  __bf16* wvT = (__bf16*)(ws + 26 * MB);    //  2 MiB
  __bf16* woT = (__bf16*)(ws + 28 * MB);    //  8 MiB
  float*  qf  = (float*) (ws + 36 * MB);    // 32 MiB (f32 Q pre-RoPE)
  float*  kf  = (float*) (ws + 68 * MB);    //  8 MiB
  float*  vf  = (float*) (ws + 76 * MB);    //  8 MiB
  __bf16* qbb = (__bf16*)(ws + 84 * MB);    // 16 MiB
  __bf16* kbb = (__bf16*)(ws + 100 * MB);   //  4 MiB
  __bf16* vbb = (__bf16*)(ws + 104 * MB);   //  4 MiB  (total 108 MiB)
  __bf16* ao  = (__bf16*)qf;                // reuse qf region after RoPE

  const int M = B_ * S_;  // 4096

  // 1) cast x -> bf16
  {
    int n = M * D_;
    cast_bf16_kernel<<<(n + 255) / 256, 256, 0, stream>>>(x, xb, n);
  }
  // 2) cast+transpose weights
  transpose_cast_kernel<<<dim3(D_ / 32, D_ / 32), dim3(32, 8), 0, stream>>>(wq, wqT, D_, D_);
  transpose_cast_kernel<<<dim3((KVH_ * HD_) / 32, D_ / 32), dim3(32, 8), 0, stream>>>(wk, wkT, D_, KVH_ * HD_);
  transpose_cast_kernel<<<dim3((KVH_ * HD_) / 32, D_ / 32), dim3(32, 8), 0, stream>>>(wv, wvT, D_, KVH_ * HD_);
  transpose_cast_kernel<<<dim3(D_ / 32, D_ / 32), dim3(32, 8), 0, stream>>>(wo, woT, D_, D_);

  // 3) QKV projections (async-LDS staged WMMA)
  gemm_bt_kernel<true><<<dim3(M / 128, D_ / 64), 256, 0, stream>>>(xb, wqT, qf, M, D_, D_);
  gemm_bt_kernel<true><<<dim3(M / 128, (KVH_ * HD_) / 64), 256, 0, stream>>>(xb, wkT, kf, M, KVH_ * HD_, D_);
  gemm_bt_kernel<true><<<dim3(M / 128, (KVH_ * HD_) / 64), 256, 0, stream>>>(xb, wvT, vf, M, KVH_ * HD_, D_);

  // 4) RoPE + layout (Q, K), transpose V
  {
    int nq = B_ * S_ * H_ * (HD_ / 2);
    rope_reformat_kernel<<<(nq + 255) / 256, 256, 0, stream>>>(qf, cosT, sinT, qbb, H_, nq);
    int nk = B_ * S_ * KVH_ * (HD_ / 2);
    rope_reformat_kernel<<<(nk + 255) / 256, 256, 0, stream>>>(kf, cosT, sinT, kbb, KVH_, nk);
    int nv = B_ * S_ * KVH_ * HD_;
    vtrans_kernel<<<(nv + 255) / 256, 256, 0, stream>>>(vf, vbb, nv);
  }

  // 5) causal GQA flash attention (WMMA)
  attn_kernel<<<dim3(S_ / 64, B_ * H_), 128, 0, stream>>>(qbb, kbb, vbb, ao);

  // 6) output projection (async-LDS staged WMMA, f32 out)
  gemm_bt_kernel<true><<<dim3(M / 128, D_ / 64), 256, 0, stream>>>(ao, woT, (float*)d_out, M, D_, D_);
}
